// GrowingSparseSNN_23055384445878
// MI455X (gfx1250) — matrix operand order repaired and checked by reading, not compiled, not run on hardware
//
#include <hip/hip_runtime.h>
#include <hip/hip_bf16.h>
#include <stdint.h>

// ---------------------------------------------------------------------------
// SNN forward: h = x @ w_in^T -> LIF scan over T -> out = s_sum @ w_out^T / T
// gfx1250: bf16 WMMA + Tensor Data Mover (TDM) double-buffered LDS staging.
// ---------------------------------------------------------------------------

#define SNN_B 256
#define SNN_T 32
#define SNN_F 2048
#define SNN_H 2048
#define SNN_O 256

#define KC    128                 // K elements staged per pipeline stage
#define NKC   (SNN_F / KC)        // 16 stages per time step
#define NSTG  (SNN_T * NKC)       // 512 total pipeline stages

typedef __attribute__((ext_vector_type(16))) __bf16 v16bf;
typedef __attribute__((ext_vector_type(8)))  float  v8f;
typedef unsigned int u32x4 __attribute__((ext_vector_type(4)));
typedef int          i32x8 __attribute__((ext_vector_type(8)));
typedef int          i32x4 __attribute__((ext_vector_type(4)));

union FragBF {
  uint4 u[2];   // 32 bytes
  v16bf v;      // 16 bf16 per lane
};

__device__ __forceinline__ unsigned rfl(unsigned v) {
  return (unsigned)__builtin_amdgcn_readfirstlane((int)v);
}

// round-to-nearest-even f32 -> bf16 (raw u16)
__device__ __forceinline__ unsigned short f2bf(float f) {
  unsigned int u = __float_as_uint(f);
  u += 0x7FFFu + ((u >> 16) & 1u);
  return (unsigned short)(u >> 16);
}

// ---------------------------------------------------------------------------
// TDM 2-D tile load: 8 rows x KC bf16 elements -> LDS (compacted row-major).
// D# per CDNA5 ISA ch.8: group0 {count, lds_addr, global_addr, type=2},
// group1 {data_size=2B, tensor dims, tile dims, dim0 stride}. Groups 2/3
// zero => 2-D mode. One issue per wave; tracked by TENSORcnt.
// ---------------------------------------------------------------------------
__device__ __forceinline__ void tdm_load_2d(unsigned lds_addr,
                                            unsigned ga_lo, unsigned ga_hi,
                                            unsigned row_stride_elems) {
  u32x4 g0;
  g0.x = 1u;                                            // count=1, user mode
  g0.y = lds_addr;                                      // LDS byte address
  g0.z = ga_lo;                                         // global addr [31:0]
  g0.w = (ga_hi & 0x01FFFFFFu) | (2u << 30);            // addr[56:32] | type=2
  i32x8 g1;
  g1.s0 = (int)(1u << 16);                              // data_size = 2 bytes
  g1.s1 = (int)((unsigned)SNN_F << 16);                 // tensor_dim0 lo16 (=2048)
  g1.s2 = (int)(0xFFFFu << 16);                         // dim0 hi=0, tensor_dim1 lo16
  g1.s3 = (int)((unsigned)KC << 16);                    // dim1 hi=0, tile_dim0 = KC
  g1.s4 = 8;                                            // tile_dim1 = 8 rows
  g1.s5 = (int)row_stride_elems;                        // tensor_dim0_stride lo32
  g1.s6 = 0;
  g1.s7 = 0;
  i32x4 z4 = {0, 0, 0, 0};
#if __has_include(<hip/amd_detail/amd_gfx1250_TDM.h>)
  i32x8 z8 = {0, 0, 0, 0, 0, 0, 0, 0};
  __builtin_amdgcn_tensor_load_to_lds(g0, g1, z4, z4, z8, 0);
#else
  __builtin_amdgcn_tensor_load_to_lds(g0, g1, z4, z4, 0);
#endif
}

// ---------------------------------------------------------------------------
// Kernel 1: fp32 -> bf16 conversion, 8 elements per thread
// ---------------------------------------------------------------------------
__global__ void snn_cvt_bf16(const float* __restrict__ src,
                             unsigned short* __restrict__ dst, int n8) {
  int idx = blockIdx.x * blockDim.x + threadIdx.x;
  if (idx >= n8) return;
  float4 a = ((const float4*)src)[2 * idx + 0];
  float4 b = ((const float4*)src)[2 * idx + 1];
  union { unsigned short h[8]; uint4 u; } o;
  o.h[0] = f2bf(a.x); o.h[1] = f2bf(a.y); o.h[2] = f2bf(a.z); o.h[3] = f2bf(a.w);
  o.h[4] = f2bf(b.x); o.h[5] = f2bf(b.y); o.h[6] = f2bf(b.z); o.h[7] = f2bf(b.w);
  ((uint4*)dst)[idx] = o.u;
}

// ---------------------------------------------------------------------------
// Kernel 2: fused GEMM + LIF. Block = 256 threads (8 waves), tile 64(b)x64(h).
// TDM double-buffered staging: stage st+1 DMAs while stage st computes.
// Each wave DMAs its own 8-row slices of the A and B tiles (2 tensor ops /
// stage / wave), then s_wait_tensorcnt(2) + barrier makes both tiles visible.
// LIF state (v, i, s_sum) lives in VGPRs across all 32 time steps.
// ---------------------------------------------------------------------------
__global__ __launch_bounds__(256)
void snn_lif_fused(const unsigned short* __restrict__ xb,   // [B][T][F] bf16
                   const unsigned short* __restrict__ wb,   // [H][F]    bf16
                   unsigned short* __restrict__ s_bf)       // [B][H]    bf16
{
  __shared__ __align__(16) unsigned short As[2][64 * KC];   // 2 x 16 KB
  __shared__ __align__(16) unsigned short Bs[2][64 * KC];   // 2 x 16 KB

  const int tid  = threadIdx.x;
  const int lane = tid & 31;
  const int wv   = tid >> 5;
  const int b0   = blockIdx.y * 64;
  const int h0   = blockIdx.x * 64;

  const int mi  = wv & 3;          // b-subtile within tile
  const int nj0 = (wv >> 2) * 2;   // first of two h-subtiles
  const int wrow = wv * 8;         // this wave's DMA row slice

  // fragment addressing (contiguous k = 16h + s, same order in A and B)
  const int arow  = mi * 16 + (lane & 15);
  const int ak0   = (lane >= 16) ? 16 : 0;    // A: K 0-15 vs 16-31
  const int brow0 = nj0 * 16 + (lane & 15);
  const int brow1 = brow0 + 16;
  const int bk0   = (lane >= 16) ? 16 : 0;    // B: K 0-15 vs 16-31

  // uniform (per wave) LDS byte addresses of this wave's DMA destinations
  unsigned ldsA[2], ldsB[2];
  ldsA[0] = rfl((unsigned)(uintptr_t)&As[0][wrow * KC]);
  ldsA[1] = rfl((unsigned)(uintptr_t)&As[1][wrow * KC]);
  ldsB[0] = rfl((unsigned)(uintptr_t)&Bs[0][wrow * KC]);
  ldsB[1] = rfl((unsigned)(uintptr_t)&Bs[1][wrow * KC]);

  const unsigned long long xbase = (unsigned long long)(uintptr_t)xb;
  const unsigned long long wbase = (unsigned long long)(uintptr_t)wb;

  auto issue = [&](int stage) {
    const int t  = stage >> 4;          // stage / NKC
    const int k0 = (stage & (NKC - 1)) * KC;
    const int bi = stage & 1;
    unsigned long long ga = xbase +
        2ull * ((((unsigned long long)(b0 + wrow)) * SNN_T + (unsigned)t) * SNN_F + (unsigned)k0);
    unsigned long long gb = wbase +
        2ull * (((unsigned long long)(h0 + wrow)) * SNN_F + (unsigned)k0);
    tdm_load_2d(ldsA[bi], rfl((unsigned)ga), rfl((unsigned)(ga >> 32)),
                (unsigned)(SNN_T * SNN_F));           // x row stride = T*F
    tdm_load_2d(ldsB[bi], rfl((unsigned)gb), rfl((unsigned)(gb >> 32)),
                (unsigned)SNN_F);                     // w_in row stride = F
  };

  v8f vm0 = {}, vm1 = {};   // membrane v
  v8f cu0 = {}, cu1 = {};   // synaptic current i
  v8f ss0 = {}, ss1 = {};   // spike sum
  v8f ha  = {}, hb  = {};   // h_t accumulators

  const float kDV = 0.001f / 20.0f;          // DT * TAU_MEM_INV
  const float kDI = 1.0f - 0.001f / 10.0f;   // 1 - DT * TAU_SYN_INV

  issue(0);  // prologue: fill buffer 0

  for (int st = 0; st < NSTG; ++st) {
    if (st + 1 < NSTG) {
      issue(st + 1);                          // prefetch into alternate buffer
      __builtin_amdgcn_s_wait_tensorcnt(2);   // in-order => stage st landed
    } else {
      __builtin_amdgcn_s_wait_tensorcnt(0);
    }
    __syncthreads();                          // tiles for stage st visible

    const unsigned short* Ash = As[st & 1];
    const unsigned short* Bsh = Bs[st & 1];
#pragma unroll
    for (int ks = 0; ks < KC / 32; ++ks) {
      const int ko = ks * 32;
      FragBF a, p, q;
      a.u[0] = *(const uint4*)&Ash[arow * KC + ko + ak0];
      a.u[1] = *(const uint4*)&Ash[arow * KC + ko + ak0 + 8];
      p.u[0] = *(const uint4*)&Bsh[brow0 * KC + ko + bk0];
      p.u[1] = *(const uint4*)&Bsh[brow0 * KC + ko + bk0 + 8];
      q.u[0] = *(const uint4*)&Bsh[brow1 * KC + ko + bk0];
      q.u[1] = *(const uint4*)&Bsh[brow1 * KC + ko + bk0 + 8];
      ha = __builtin_amdgcn_wmma_f32_16x16x32_bf16(false, a.v, false, p.v,
                                                   (short)0, ha, false, false);
      hb = __builtin_amdgcn_wmma_f32_16x16x32_bf16(false, a.v, false, q.v,
                                                   (short)0, hb, false, false);
    }

    if ((st & (NKC - 1)) == NKC - 1) {
      // end of time step: LIF update (norse lif_step; old i decays v, then h_t injects)
#pragma unroll
      for (int e = 0; e < 8; ++e) {
        float vd0 = vm0[e] + kDV * (cu0[e] - vm0[e]);
        float vd1 = vm1[e] + kDV * (cu1[e] - vm1[e]);
        float z0 = (vd0 > 1.0f) ? 1.0f : 0.0f;
        float z1 = (vd1 > 1.0f) ? 1.0f : 0.0f;
        ss0[e] += z0;
        ss1[e] += z1;
        vm0[e] = (1.0f - z0) * vd0;           // V_RESET = 0
        vm1[e] = (1.0f - z1) * vd1;
        cu0[e] = cu0[e] * kDI + ha[e];
        cu1[e] = cu1[e] * kDI + hb[e];
        ha[e] = 0.0f;
        hb[e] = 0.0f;
      }
    }
    __syncthreads();   // stage st fully consumed; its buffer is DMA-refillable
  }

  // write spike counts (small integers -> exact in bf16)
  const int n0 = h0 + nj0 * 16 + (lane & 15);
  const int mb = (lane >= 16) ? 8 : 0;
#pragma unroll
  for (int r = 0; r < 8; ++r) {
    const int m = b0 + mi * 16 + r + mb;
    s_bf[(size_t)m * SNN_H + n0]      = f2bf(ss0[r]);
    s_bf[(size_t)m * SNN_H + n0 + 16] = f2bf(ss1[r]);
  }
}

// ---------------------------------------------------------------------------
// Kernel 3: out[b][o] = (s_sum @ w_out^T)[b][o] / 32.  One wave per 16x16 tile.
// 256 subtiles -> 32 blocks x 8 waves. Operands ~2 MB, L2-resident.
// ---------------------------------------------------------------------------
__global__ __launch_bounds__(256)
void snn_out_gemm(const unsigned short* __restrict__ s_bf,  // [B][H] bf16
                  const unsigned short* __restrict__ wob,   // [O][H] bf16
                  float* __restrict__ out)                  // [B][O] f32
{
  const int lane = threadIdx.x & 31;
  const int wv   = (blockIdx.x * blockDim.x + threadIdx.x) >> 5;  // 0..255
  const int sm   = (wv >> 4) * 16;   // b tile
  const int sn   = (wv & 15) * 16;   // o tile

  const int arow = sm + (lane & 15);
  const int ak0  = (lane >= 16) ? 16 : 0;
  const int brow = sn + (lane & 15);
  const int bk0  = (lane >= 16) ? 16 : 0;

  v8f acc = {};
  for (int kc = 0; kc < SNN_H / 32; ++kc) {
    const int k0 = kc * 32;
    FragBF a, b;
    a.u[0] = *(const uint4*)&s_bf[(size_t)arow * SNN_H + k0 + ak0];
    a.u[1] = *(const uint4*)&s_bf[(size_t)arow * SNN_H + k0 + ak0 + 8];
    b.u[0] = *(const uint4*)&wob[(size_t)brow * SNN_H + k0 + bk0];
    b.u[1] = *(const uint4*)&wob[(size_t)brow * SNN_H + k0 + bk0 + 8];
    acc = __builtin_amdgcn_wmma_f32_16x16x32_bf16(false, a.v, false, b.v,
                                                  (short)0, acc, false, false);
  }
  const int mb = (lane >= 16) ? 8 : 0;
  const int n  = sn + (lane & 15);
#pragma unroll
  for (int r = 0; r < 8; ++r) {
    out[(sm + r + mb) * SNN_O + n] = acc[r] * (1.0f / 32.0f);
  }
}

// ---------------------------------------------------------------------------
extern "C" void kernel_launch(void* const* d_in, const int* in_sizes, int n_in,
                              void* d_out, int out_size, void* d_ws, size_t ws_size,
                              hipStream_t stream) {
  const float* x     = (const float*)d_in[0];  // [B][T][F]
  const float* w_in  = (const float*)d_in[1];  // [H][F]
  const float* w_out = (const float*)d_in[2];  // [O][H]

  // workspace layout (bf16 as u16): xb 32MB | wb 8MB | wob 1MB | s 1MB = 42MB
  unsigned short* xb  = (unsigned short*)d_ws;
  unsigned short* wb  = xb  + (size_t)SNN_B * SNN_T * SNN_F;
  unsigned short* wob = wb  + (size_t)SNN_H * SNN_F;
  unsigned short* sbf = wob + (size_t)SNN_O * SNN_H;

  const int nx8 = SNN_B * SNN_T * SNN_F / 8;
  const int nw8 = SNN_H * SNN_F / 8;
  const int no8 = SNN_O * SNN_H / 8;
  snn_cvt_bf16<<<nx8 / 256, 256, 0, stream>>>(x, xb, nx8);
  snn_cvt_bf16<<<nw8 / 256, 256, 0, stream>>>(w_in, wb, nw8);
  snn_cvt_bf16<<<no8 / 256, 256, 0, stream>>>(w_out, wob, no8);

  dim3 g2(SNN_H / 64, SNN_B / 64);  // 32 x 4 = 128 workgroups
  snn_lif_fused<<<g2, 256, 0, stream>>>(xb, wb, sbf);

  snn_out_gemm<<<32, 256, 0, stream>>>(sbf, wob, (float*)d_out);
}
